// GateLoopedAttention_75265006895296
// MI455X (gfx1250) — matrix operand order repaired and checked by reading, not compiled, not run on hardware
//
#include <hip/hip_runtime.h>
#include <hip/hip_bf16.h>
#include <math.h>

// ---------------------------------------------------------------------------
// GateLoopedAttention for MI455X (gfx1250, wave32, WMMA + double-buffered TDM)
// B=4, N=8192, DIM=512, H=512, DH=1  ->  scan state is scalar per (b,h).
// ---------------------------------------------------------------------------

#define Bb   4
#define Nn   8192
#define DIMc 512
#define Hh   512
#define Mrows (Bb * Nn)          // 32768
#define Ocat  3072               // 1536 (qkv) + 1024 (a) + 512 (g)
#define Kdim  512

// LDS staging: 64 weight rows x 128 K-halfs per stage, TDM pads 4 dwords after
// every 64 dwords (one 256B row) -> LDS row stride 136 halfs (272B = 68 dwords;
// 68 % 64 = 4 -> 4-bank rotation/row, 16 lanes x 4 banks = all 64 banks).
#define KSTAGE 128
#define BROW   136
#define BUFH   (64 * BROW)       // halfs per buffer (17408B)

typedef __attribute__((ext_vector_type(16))) _Float16 v16h;
typedef __attribute__((ext_vector_type(8)))  _Float16 v8h;
typedef __attribute__((ext_vector_type(8)))  float    v8f;
typedef __attribute__((ext_vector_type(4)))  unsigned int u32x4;
typedef __attribute__((ext_vector_type(8)))  int      i32x8;
typedef __attribute__((ext_vector_type(4)))  int      i32x4;

__device__ __forceinline__ float sigmoidf_(float x) { return 1.0f / (1.0f + __expf(-x)); }

// ---------------------------------------------------------------------------
// 1) RMSNorm: xn = x / max(||x||,1e-12) * sqrt(512) * gamma   (f16 out)
// ---------------------------------------------------------------------------
__global__ __launch_bounds__(128) void rmsnorm_kernel(const float* __restrict__ x,
                                                      const float* __restrict__ gamma,
                                                      _Float16* __restrict__ xn16) {
    const int row = blockIdx.x;
    const int t = threadIdx.x;
    const float* xr = x + (size_t)row * DIMc;
    float vals[4];
    float ss = 0.f;
    #pragma unroll
    for (int i = 0; i < 4; ++i) { vals[i] = xr[t + 128 * i]; ss += vals[i] * vals[i]; }
    #pragma unroll
    for (int off = 16; off > 0; off >>= 1) ss += __shfl_xor(ss, off, 32);
    __shared__ float wss[4];
    if ((t & 31) == 0) wss[t >> 5] = ss;
    __syncthreads();
    ss = wss[0] + wss[1] + wss[2] + wss[3];
    const float inv = 22.62741699796952f / fmaxf(sqrtf(ss), 1e-12f); // sqrt(512)/||x||
    _Float16* out = xn16 + (size_t)row * DIMc;
    #pragma unroll
    for (int i = 0; i < 4; ++i) {
        const int c = t + 128 * i;
        out[c] = (_Float16)(vals[i] * inv * gamma[c]);
    }
}

// ---------------------------------------------------------------------------
// 2) Pack weights to f16. Wcat = [w_qkv ; w_a ; w_g] (3072 x 512), Wo (512x512)
// ---------------------------------------------------------------------------
__global__ __launch_bounds__(256) void pack_weights_kernel(const float* __restrict__ w_qkv,
                                                           const float* __restrict__ w_a,
                                                           const float* __restrict__ w_g,
                                                           const float* __restrict__ w_o,
                                                           _Float16* __restrict__ wcat16,
                                                           _Float16* __restrict__ wo16) {
    const size_t idx = (size_t)blockIdx.x * 256 + threadIdx.x;
    const size_t ncat = (size_t)Ocat * Kdim;
    const size_t nwo  = (size_t)DIMc * DIMc;
    if (idx < ncat) {
        const int o = (int)(idx / Kdim);
        const int k = (int)(idx % Kdim);
        float v;
        if (o < 1536)      v = w_qkv[(size_t)o * Kdim + k];
        else if (o < 2560) v = w_a[(size_t)(o - 1536) * Kdim + k];
        else               v = w_g[(size_t)(o - 2560) * Kdim + k];
        wcat16[idx] = (_Float16)v;
    } else if (idx < ncat + nwo) {
        const size_t j = idx - ncat;
        wo16[j] = (_Float16)w_o[j];
    }
}

// ---------------------------------------------------------------------------
// TDM issue (no wait): stage a 64-row x KSTAGE-half block of W (row-major,
// row stride Kdim halfs) into LDS at byte offset ldsOff; pad 4 dwords per
// 64 dwords so LDS row stride = BROW halfs.  D# per CDNA5 ISA sec. 8.3/8.4.
// ---------------------------------------------------------------------------
__device__ __forceinline__ void tdm_issue_block(const _Float16* src, unsigned ldsOff) {
    const unsigned long long ga = (unsigned long long)(__SIZE_TYPE__)src;

    u32x4 g0;
    g0[0] = 1u;                                   // count=1, user descriptor
    g0[1] = ldsOff;                               // lds_addr (bytes)
    g0[2] = (unsigned)(ga & 0xFFFFFFFFu);         // global_addr[31:0]
    g0[3] = (unsigned)((ga >> 32) & 0x01FFFFFFu)  // global_addr[56:32]
          | (2u << 30);                           // type = 2 ("image")

    i32x8 g1;
    g1[0] = (1 << 16)                             // data_size = 2 bytes
          | (1 << 20)                             // pad_enable
          | (5 << 22)                             // pad_interval: 64 dwords (256B)
          | (3 << 25);                            // pad_amount: 4 dwords (16B)
    g1[1] = (Kdim << 16);                         // tensor_dim0 = 512 (lo16)
    g1[2] = (Ocat << 16);                         // tensor_dim0 hi=0; tensor_dim1 lo16
    g1[3] = (KSTAGE << 16);                       // tensor_dim1 hi=0; tile_dim0 = 128
    g1[4] = 64;                                   // tile_dim1 = 64; tile_dim2 = 0
    g1[5] = Kdim;                                 // tensor_dim0_stride = 512 (lo32)
    g1[6] = 0;
    g1[7] = 0;
    i32x4 z4 = {0, 0, 0, 0};
    i32x8 z8 = {0, 0, 0, 0, 0, 0, 0, 0};

    __builtin_amdgcn_tensor_load_to_lds(g0, g1, z4, z4, z8, 0);
}

// ---------------------------------------------------------------------------
// 3) WMMA GEMM: Y[M,O] = A[M,K] (f16) * W[O,K]^T (f16), fp32 accumulate.
// Workgroup = 8 waves; block tile = 128 M-rows x 64 O-cols.
// TDM double-buffers 64xKSTAGE weight blocks into LDS (issued by wave 0,
// stage kh+1 prefetched while computing stage kh); waves read B fragments
// from LDS; each wave owns one 16-row M-tile and 4 O-subtiles, so one A
// fragment feeds 4 v_wmma_f32_16x16x32_f16 per K-step.
// A fragment (16-bit 16x32 layout): lane<16 holds K 0..7 & 16..23 of row
// (lane&15); lane>=16 holds K 8..15 & 24..31.  B fragment: lane holds column
// (lane&15), same K order as A (K (lane>>4)*8..+7 & 16+(lane>>4)*8..+7).
// A dummy store into the TDM padding region makes the DMA's LDS writes
// visible to the compiler (otherwise B loads get undef-folded/hoisted).
// mode 0: fused epilogue (qkv f16 | a + b_a | silu gates); mode 1: fp32 out.
// ---------------------------------------------------------------------------
__global__ __launch_bounds__(256) void wmma_gemm_kernel(const _Float16* __restrict__ A,
                                                        const _Float16* __restrict__ W,
                                                        int M, int O, int mode,
                                                        _Float16* __restrict__ qkv16,
                                                        float* __restrict__ aproj,
                                                        const float* __restrict__ b_a,
                                                        float* __restrict__ gates,
                                                        float* __restrict__ outf) {
    __shared__ _Float16 bsm[2 * BUFH];            // 2 x 17 KB double buffer

    const int lane = threadIdx.x & 31;
    const int wave = threadIdx.x >> 5;
    const int tilesO64 = O >> 6;
    const int bm = blockIdx.x / tilesO64;
    const int bo = blockIdx.x % tilesO64;
    const int m0 = bm * 128 + wave * 16;
    const int o0 = bo * 64;

    const int r  = lane & 15;
    const int hi = lane >> 4;

    const _Float16* arow = A + (size_t)(m0 + r) * Kdim;
    const _Float16* wblk = W + (size_t)o0 * Kdim;

    v8f acc[4];
    #pragma unroll
    for (int t = 0; t < 4; ++t) acc[t] = (v8f){};

    const int NK = Kdim / KSTAGE;                 // 4 stages
    if (wave == 0)
        tdm_issue_block(wblk, 0);                 // stage 0 -> buffer 0

    for (int kh = 0; kh < NK; ++kh) {
        const int buf = kh & 1;
        if (wave == 0)
            __builtin_amdgcn_s_wait_tensorcnt(0); // buffer `buf` ready
        if (threadIdx.x == 0)
            bsm[buf * BUFH + KSTAGE] = (_Float16)(float)kh;  // pad slot: visible store
        asm volatile("" ::: "memory");
        __syncthreads();                          // staged data visible to all waves

        if (wave == 0 && kh + 1 < NK)             // prefetch next stage (async)
            tdm_issue_block(wblk + (kh + 1) * KSTAGE, (unsigned)((buf ^ 1) * BUFH * 2));

        const _Float16* bbase = bsm + buf * BUFH;
        #pragma unroll
        for (int kk = 0; kk < KSTAGE; kk += 32) {
            const int kb = kh * KSTAGE + kk;
            v8h alo = *(const v8h*)(arow + kb + hi * 8);
            v8h ahi = *(const v8h*)(arow + kb + 16 + hi * 8);
            v16h a = __builtin_shufflevector(alo, ahi,
                     0, 1, 2, 3, 4, 5, 6, 7, 8, 9, 10, 11, 12, 13, 14, 15);
            #pragma unroll
            for (int t = 0; t < 4; ++t) {
                const _Float16* bp = bbase + (t * 16 + r) * BROW + kk + hi * 8;
                v8h b0 = *(const v8h*)(bp);
                v8h b1 = *(const v8h*)(bp + 16);
                v16h b = __builtin_shufflevector(b0, b1,
                         0, 1, 2, 3, 4, 5, 6, 7, 8, 9, 10, 11, 12, 13, 14, 15);
                acc[t] = __builtin_amdgcn_wmma_f32_16x16x32_f16(
                    false, a, false, b, (short)0, acc[t], false, false);
            }
        }
        __syncthreads();                          // done reading before next overwrite
    }

    // C layout: VGPR i -> row hi*8+i, col = lane&15
    const int rbase = m0 + hi * 8;
    #pragma unroll
    for (int t = 0; t < 4; ++t) {
        const int c = o0 + t * 16 + r;
        if (mode == 0) {
            if (c < 1536) {
                #pragma unroll
                for (int i = 0; i < 8; ++i)
                    qkv16[(size_t)(rbase + i) * 1536 + c] = (_Float16)acc[t][i];
            } else if (c < 2560) {
                const float bias = b_a[c - 1536];
                #pragma unroll
                for (int i = 0; i < 8; ++i)
                    aproj[(size_t)(rbase + i) * 1024 + (c - 1536)] = acc[t][i] + bias;
            } else {
                #pragma unroll
                for (int i = 0; i < 8; ++i) {
                    const float v = acc[t][i];
                    gates[(size_t)(rbase + i) * 512 + (c - 2560)] = v * sigmoidf_(v);
                }
            }
        } else {
            #pragma unroll
            for (int i = 0; i < 8; ++i)
                outf[(size_t)(rbase + i) * O + c] = acc[t][i];
        }
    }
}

// ---------------------------------------------------------------------------
// 4) Gate prep + transpose: g[bh][n] = sigmoid(|a|)*ar/|a| ; kT/vT planes.
// (polar(sigmoid(mag), phase).real ; only the real part feeds the scan.)
// ---------------------------------------------------------------------------
__global__ __launch_bounds__(256) void gate_prep_kernel(const float* __restrict__ aproj,
                                                        const _Float16* __restrict__ qkv16,
                                                        float* __restrict__ g,
                                                        _Float16* __restrict__ kT,
                                                        _Float16* __restrict__ vT) {
    const size_t idx = (size_t)blockIdx.x * 256 + threadIdx.x; // over B*N*H
    if (idx >= (size_t)Mrows * Hh) return;
    const int h = (int)(idx % Hh);
    const size_t m = idx / Hh;              // m = b*N + n
    const int n = (int)(m % Nn);
    const int b = (int)(m / Nn);

    const float ar = aproj[m * 1024 + 2 * h];
    const float ai = aproj[m * 1024 + 2 * h + 1];
    const float mag = sqrtf(ar * ar + ai * ai);
    const float s = sigmoidf_(mag);
    const float gr = (mag > 0.f) ? s * (ar / mag) : s;   // angle(0)=0 -> cos=1

    const size_t plane = ((size_t)b * Hh + h) * Nn + n;
    g[plane] = gr;
    kT[plane] = qkv16[m * 1536 + 512 + h];
    vT[plane] = qkv16[m * 1536 + 1024 + h];
}

// ---------------------------------------------------------------------------
// 5) Linear-recurrence scan: kv_n = g_n * kv_{n-1} + k_n*v_n.
// One wave per (b,h) sequence; 32-wide shuffle scan per chunk, serial carry.
// ---------------------------------------------------------------------------
__global__ __launch_bounds__(256) void scan_kernel(const float* __restrict__ g,
                                                   const _Float16* __restrict__ kT,
                                                   const _Float16* __restrict__ vT,
                                                   float* __restrict__ kvT) {
    const int lane = threadIdx.x & 31;
    const int bh = blockIdx.x * 8 + (threadIdx.x >> 5);   // 0..2047
    const size_t base = (size_t)bh * Nn;

    float carry = 0.f;
    for (int n0 = 0; n0 < Nn; n0 += 32) {
        float gg = g[base + n0 + lane];
        float u = (float)kT[base + n0 + lane] * (float)vT[base + n0 + lane];
        #pragma unroll
        for (int off = 1; off < 32; off <<= 1) {
            const float gp = __shfl_up(gg, off, 32);
            const float up = __shfl_up(u, off, 32);
            if (lane >= off) { u = fmaf(gg, up, u); gg *= gp; }
        }
        const float kv = fmaf(gg, carry, u);
        kvT[base + n0 + lane] = kv;
        carry = __shfl(kv, 31, 32);
    }
}

// ---------------------------------------------------------------------------
// 6) Merge: G16[m][d] = (f16)(gates * q * kv)   (input to the final GEMM)
// ---------------------------------------------------------------------------
__global__ __launch_bounds__(256) void merge_kernel(const float* __restrict__ kvT,
                                                    const _Float16* __restrict__ qkv16,
                                                    const float* __restrict__ gates,
                                                    _Float16* __restrict__ G16) {
    const size_t idx = (size_t)blockIdx.x * 256 + threadIdx.x; // over M*DIM
    if (idx >= (size_t)Mrows * DIMc) return;
    const int d = (int)(idx % DIMc);
    const size_t m = idx / DIMc;
    const int n = (int)(m % Nn);
    const int b = (int)(m / Nn);
    const float q = (float)qkv16[m * 1536 + d];
    const float kv = kvT[((size_t)b * Hh + d) * Nn + n];
    G16[idx] = (_Float16)(gates[idx] * q * kv);
}

// ---------------------------------------------------------------------------
// Launch
// ---------------------------------------------------------------------------
static inline size_t align256(size_t x) { return (x + 255) & ~(size_t)255; }

extern "C" void kernel_launch(void* const* d_in, const int* in_sizes, int n_in,
                              void* d_out, int out_size, void* d_ws, size_t ws_size,
                              hipStream_t stream) {
    const float* x     = (const float*)d_in[0];
    const float* gamma = (const float*)d_in[1];
    const float* w_qkv = (const float*)d_in[2];
    const float* w_a   = (const float*)d_in[3];
    const float* b_a   = (const float*)d_in[4];
    const float* w_g   = (const float*)d_in[5];
    const float* w_o   = (const float*)d_in[6];
    float* out = (float*)d_out;

    char* ws = (char*)d_ws;
    size_t off = 0;
    auto alloc = [&](size_t bytes) { char* p = ws + off; off += align256(bytes); return p; };

    _Float16* xn16   = (_Float16*)alloc((size_t)Mrows * DIMc * 2);   //  32 MB
    _Float16* wcat16 = (_Float16*)alloc((size_t)Ocat * Kdim * 2);    //   3 MB
    _Float16* wo16   = (_Float16*)alloc((size_t)DIMc * DIMc * 2);    // 0.5 MB
    _Float16* qkv16  = (_Float16*)alloc((size_t)Mrows * 1536 * 2);   //  96 MB
    float*    aproj  = (float*)   alloc((size_t)Mrows * 1024 * 4);   // 128 MB
    float*    gates  = (float*)   alloc((size_t)Mrows * 512 * 4);    //  64 MB
    float*    g      = (float*)   alloc((size_t)Bb * Hh * Nn * 4);   //  64 MB
    _Float16* kT     = (_Float16*)alloc((size_t)Bb * Hh * Nn * 2);   //  32 MB
    _Float16* vT     = (_Float16*)alloc((size_t)Bb * Hh * Nn * 2);   //  32 MB
    float*    kvT    = (float*)   alloc((size_t)Bb * Hh * Nn * 4);   //  64 MB
    _Float16* G16    = (_Float16*)alloc((size_t)Mrows * DIMc * 2);   //  32 MB

    // 1) RMSNorm
    rmsnorm_kernel<<<Mrows, 128, 0, stream>>>(x, gamma, xn16);

    // 2) Pack weights (f16)
    {
        const size_t n = (size_t)Ocat * Kdim + (size_t)DIMc * DIMc;
        pack_weights_kernel<<<(int)((n + 255) / 256), 256, 0, stream>>>(
            w_qkv, w_a, w_g, w_o, wcat16, wo16);
    }

    // 3) Fused projection GEMM: [qkv | a | g] = xn @ Wcat^T  (TDM->LDS->WMMA)
    {
        const int blocks = (Mrows / 128) * (Ocat / 64);   // 256 * 48 = 12288
        wmma_gemm_kernel<<<blocks, 256, 0, stream>>>(
            xn16, wcat16, Mrows, Ocat, /*mode=*/0,
            qkv16, aproj, b_a, gates, nullptr);
    }

    // 4) Gate prep + k/v transpose
    {
        const size_t n = (size_t)Mrows * Hh;
        gate_prep_kernel<<<(int)((n + 255) / 256), 256, 0, stream>>>(
            aproj, qkv16, g, kT, vT);
    }

    // 5) Scan (2048 sequences, 8 waves/block)
    scan_kernel<<<(Bb * Hh) / 8, 256, 0, stream>>>(g, kT, vT, kvT);

    // 6) Merge to f16
    {
        const size_t n = (size_t)Mrows * DIMc;
        merge_kernel<<<(int)((n + 255) / 256), 256, 0, stream>>>(kvT, qkv16, gates, G16);
    }

    // 7) Final GEMM: out = G16 @ w_o^T  (TDM->LDS->WMMA)
    {
        const int blocks = (Mrows / 128) * (DIMc / 64);   // 256 * 8 = 2048
        wmma_gemm_kernel<<<blocks, 256, 0, stream>>>(
            G16, wo16, Mrows, DIMc, /*mode=*/1,
            nullptr, nullptr, nullptr, nullptr, out);
    }
}
